// RelPosSelfAttention_3075196584198
// MI455X (gfx1250) — matrix operand order repaired and checked by reading, not compiled, not run on hardware
//
#include <hip/hip_runtime.h>
#include <stdint.h>

typedef __bf16 bf16;
typedef __attribute__((ext_vector_type(16))) __bf16 v16bf;
typedef __attribute__((ext_vector_type(8)))  __bf16 v8bf;
typedef __attribute__((ext_vector_type(4)))  __bf16 v4bf;
typedef __attribute__((ext_vector_type(2)))  __bf16 v2bf;
typedef __attribute__((ext_vector_type(8)))  float  v8f;
typedef __attribute__((ext_vector_type(4)))  float  v4f;
typedef __attribute__((ext_vector_type(4)))  unsigned int v4u;

#define LOG2E 1.44269504088896340736f

namespace {
constexpr int NH  = 8;
constexpr int L   = 1024;
constexpr int D   = 64;
constexpr int QT  = 64;      // query rows per block
constexpr int KTW = 64;      // keys per tile
constexpr int NT  = L / KTW; // 16 key tiles
constexpr int ST  = 72;      // LDS bf16 row stride in halves (144 B)

constexpr int SM_QS = 0;       // bf16 [64][72]  scaled Q tile
constexpr int SM_K0 = 9216;    // bf16 [64][72]  K tile buf0 (phase A: rel_emb_w)
constexpr int SM_K1 = 18432;   // bf16 [64][72]  K tile buf1
constexpr int SM_V0 = 27648;   // bf16 [64][72]  V^T tile buf0 (phase A: rel_emb_h)
constexpr int SM_V1 = 36864;   // bf16 [64][72]  V^T tile buf1
constexpr int SM_BW = 46080;   // f32 [64][32]   bias over key-col q
constexpr int SM_BH = 54272;   // f32 [64][32]   bias over key-row p
constexpr int SM_MB = 62464;   // bf16 [1024]    mask bias
constexpr int SM_DUMP = 64512; // f32 dump word for branchless scatter
constexpr int SM_TOT  = 64528; // < 64 KB
}

__device__ __forceinline__ v16bf cat8(v8bf lo, v8bf hi) {
  return __builtin_shufflevector(lo, hi, 0,1,2,3,4,5,6,7,8,9,10,11,12,13,14,15);
}
// A-matrix 16x32 bf16 fragment: lane m<16: K[ks..ks+15]; lane m+16: K[ks+16..ks+31].
__device__ __forceinline__ v16bf frag_a(const bf16* row, int ks, int hf) {
  int o = ks + hf * 16;
  v8bf lo = *(const v8bf*)(row + o);
  v8bf hi = *(const v8bf*)(row + o + 8);
  return cat8(lo, hi);
}
// B-matrix 32x16 bf16 fragment: lane n<16: K[ks..ks+15]; lane n+16: K[ks+16..ks+31].
__device__ __forceinline__ v16bf frag_b(const bf16* row, int ks, int hf) {
  int o = ks + hf * 16;
  v8bf lo = *(const v8bf*)(row + o);
  v8bf hi = *(const v8bf*)(row + o + 8);
  return cat8(lo, hi);
}
__device__ __forceinline__ v8f wmma_bf(v16bf a, v16bf b, v8f c) {
  return __builtin_amdgcn_wmma_f32_16x16x32_bf16(false, a, false, b, (short)0, c,
                                                 false, false);
}
__device__ __forceinline__ v8f zero8() {
  v8f z = {0.f,0.f,0.f,0.f,0.f,0.f,0.f,0.f};
  return z;
}
__device__ __forceinline__ unsigned int pack2(float a, float b) {
  v2bf t; t[0] = (bf16)a; t[1] = (bf16)b;
  return __builtin_bit_cast(unsigned int, t);
}
__device__ __forceinline__ v16bf frag_from(v4u lo, v4u hi) {
  return cat8(__builtin_bit_cast(v8bf, lo), __builtin_bit_cast(v8bf, hi));
}

__launch_bounds__(128)
__global__ void relpos_attn_wmma(const float* __restrict__ qg,
                                 const float* __restrict__ kg,
                                 const float* __restrict__ vg,
                                 const unsigned char* __restrict__ maskg,
                                 const float* __restrict__ relwg,
                                 const float* __restrict__ relhg,
                                 float* __restrict__ outg) {
  __shared__ __align__(16) unsigned char smem[SM_TOT];
  bf16*  Qs  = (bf16*)(smem + SM_QS);
  bf16*  Kl0 = (bf16*)(smem + SM_K0);
  bf16*  Kl1 = (bf16*)(smem + SM_K1);
  bf16*  Vt0 = (bf16*)(smem + SM_V0);
  bf16*  Vt1 = (bf16*)(smem + SM_V1);
  float* bw  = (float*)(smem + SM_BW);
  float* bh  = (float*)(smem + SM_BH);
  bf16*  mb  = (bf16*)(smem + SM_MB);
  float* f32s = (float*)smem;            // word-indexed view for branchless scatter

  const int tid  = threadIdx.x;
  const int lane = tid & 31;
  const int wave = tid >> 5;
  const int hf   = lane >> 4;   // half of the wave
  const int ln   = lane & 15;

  const int qtile = blockIdx.x & 15;
  const int head  = (blockIdx.x >> 4) & 7;
  const int batch = blockIdx.x >> 7;

  const size_t bn = (size_t)(batch * NH + head);
  const float* qb = qg + bn * (size_t)(L * D);
  const float* kb = kg + bn * (size_t)(L * D);
  const float* vb = vg + bn * (size_t)(L * D);
  const unsigned char* mrow = maskg + batch * L;

  const int lrow = wave * 16 + ln;        // this lane's query (local)
  const int grow = qtile * QT + lrow;     // global query linear index
  const int wq = grow & 31, hq = grow >> 5;

  // V staging geometry: 4-key x 4-dim register block transpose
  const int vk0 = (tid >> 4) * 4;         // first key of block (work item 0)
  const int vc4 = tid & 15;               // dim group

  // ---- phase 0: stage scaled Q, rel embeddings (into K0/V0), mask bias ----
  {
    const v4f* q4 = (const v4f*)(qb + (size_t)qtile * QT * D);
    const v4f* w4 = (const v4f*)relwg;
    const v4f* h4 = (const v4f*)relhg;
    v4f zf = {0.f, 0.f, 0.f, 0.f};
    for (int i = tid; i < QT * 16; i += 128) {
      int row = i >> 4, c4 = i & 15;
      v4f f = q4[row * 16 + c4];
      v4bf p;
      p[0] = (bf16)(f[0] * 0.125f);
      p[1] = (bf16)(f[1] * 0.125f);
      p[2] = (bf16)(f[2] * 0.125f);
      p[3] = (bf16)(f[3] * 0.125f);
      *(v4bf*)(Qs + row * ST + c4 * 4) = p;

      v4f fw = (row < 63) ? w4[row * 16 + c4] : zf;
      v4bf pw;
      pw[0] = (bf16)fw[0]; pw[1] = (bf16)fw[1]; pw[2] = (bf16)fw[2]; pw[3] = (bf16)fw[3];
      *(v4bf*)(Kl0 + row * ST + c4 * 4) = pw;

      v4f fh = (row < 63) ? h4[row * 16 + c4] : zf;
      v4bf ph;
      ph[0] = (bf16)fh[0]; ph[1] = (bf16)fh[1]; ph[2] = (bf16)fh[2]; ph[3] = (bf16)fh[3];
      *(v4bf*)(Vt0 + row * ST + c4 * 4) = ph;
    }
    for (int i = tid; i < L; i += 128)
      mb[i] = mrow[i] ? (bf16)0.f : (bf16)(-3.0e38f);
  }
  __syncthreads();

  // Per-lane Q^T B-fragments (column = own query); live for the whole kernel.
  const bf16* qr = Qs + lrow * ST;
  const v16bf Bq0 = frag_b(qr, 0, hf);
  const v16bf Bq1 = frag_b(qr, 32, hf);

  // Issue tile-0 K/V global loads now so they overlap the phase-A WMMAs.
  v4f kst[8], vst[2][4];
  {
    const v4f* k4 = (const v4f*)kb;
    const v4f* v4p = (const v4f*)vb;
    #pragma unroll
    for (int u = 0; u < 8; ++u) kst[u] = k4[tid + u * 128];
    #pragma unroll
    for (int wi = 0; wi < 2; ++wi)
      #pragma unroll
      for (int e = 0; e < 4; ++e)
        vst[wi][e] = v4p[(vk0 + wi * 32 + e) * 16 + vc4];
  }

  // ---- phase A: REL^T = rel_emb * Q^T via WMMA, branchless shifted scatter ----
  #pragma unroll
  for (int jt = 0; jt < 4; ++jt) {
    const bf16* wr = Kl0 + (jt * 16 + ln) * ST;
    const bf16* hr = Vt0 + (jt * 16 + ln) * ST;
    v8f cw = zero8(), ch = zero8();
    cw = wmma_bf(frag_a(wr, 0, hf), Bq0, cw);
    cw = wmma_bf(frag_a(wr, 32, hf), Bq1, cw);
    ch = wmma_bf(frag_a(hr, 0, hf), Bq0, ch);
    ch = wmma_bf(frag_a(hr, 32, hf), Bq1, ch);
    #pragma unroll
    for (int r = 0; r < 8; ++r) {
      int jg = jt * 16 + r + 8 * hf;       // relative index 0..63 (63 = pad)
      int qq = jg - 31 + wq;               // abs key col
      int pp = jg - 31 + hq;               // abs key row
      bool okw = (jg < 63) && ((unsigned)qq < 32u);
      bool okh = (jg < 63) && ((unsigned)pp < 32u);
      int iw = okw ? (SM_BW / 4 + lrow * 32 + qq) : (SM_DUMP / 4);
      int ih = okh ? (SM_BH / 4 + lrow * 32 + pp) : (SM_DUMP / 4);
      f32s[iw] = cw[r];
      f32s[ih] = ch[r];
    }
  }
  // store tile 0 (bf16 K row-major, V transposed via 4x4 register blocks) into buf 1
  #pragma unroll
  for (int u = 0; u < 8; ++u) {
    int i = tid + u * 128;
    int row = i >> 4, c4 = i & 15;
    v4f f = kst[u];
    v4bf p;
    p[0] = (bf16)f[0]; p[1] = (bf16)f[1]; p[2] = (bf16)f[2]; p[3] = (bf16)f[3];
    *(v4bf*)(Kl1 + row * ST + c4 * 4) = p;
  }
  #pragma unroll
  for (int wi = 0; wi < 2; ++wi) {
    int k0 = vk0 + wi * 32;
    #pragma unroll
    for (int d = 0; d < 4; ++d) {
      v4bf pv;
      pv[0] = (bf16)vst[wi][0][d]; pv[1] = (bf16)vst[wi][1][d];
      pv[2] = (bf16)vst[wi][2][d]; pv[3] = (bf16)vst[wi][3][d];
      *(v4bf*)(Vt1 + (vc4 * 4 + d) * ST + k0) = pv;
    }
  }
  __syncthreads();

  // ---- phase B: flash attention, S^T orientation, double-buffered K/V ----
  v8f O[4];
  #pragma unroll
  for (int jd = 0; jd < 4; ++jd) O[jd] = zero8();
  float mcur = -3.0e38f, scur = 0.f;

  for (int kt = 0; kt < NT; ++kt) {
    const int cur = (kt + 1) & 1;          // tile kt lives in buffer (kt+1)&1
    const bf16* Kc = cur ? Kl1 : Kl0;
    const bf16* Vc = cur ? Vt1 : Vt0;
    bf16* Kn = cur ? Kl0 : Kl1;
    bf16* Vn = cur ? Vt0 : Vt1;
    const bool more = (kt + 1) < NT;

    // prefetch next tile into registers (latency overlapped with WMMAs below)
    if (more) {
      const v4f* k4 = (const v4f*)(kb + (size_t)(kt + 1) * KTW * D);
      const v4f* v4p = (const v4f*)(vb + (size_t)(kt + 1) * KTW * D);
      #pragma unroll
      for (int u = 0; u < 8; ++u) kst[u] = k4[tid + u * 128];
      #pragma unroll
      for (int wi = 0; wi < 2; ++wi)
        #pragma unroll
        for (int e = 0; e < 4; ++e)
          vst[wi][e] = v4p[(vk0 + wi * 32 + e) * 16 + vc4];
    }

    // S^T = K * Q^T (64 keys x 16 queries per wave)
    v8f S[4];
    #pragma unroll
    for (int jk = 0; jk < 4; ++jk) {
      const bf16* kr = Kc + (jk * 16 + ln) * ST;
      v8f c = zero8();
      c = wmma_bf(frag_a(kr, 0, hf), Bq0, c);
      c = wmma_bf(frag_a(kr, 32, hf), Bq1, c);
      S[jk] = c;
    }

    // bias + mask: contiguous per-lane vector loads, packed-f32 vector adds
    #pragma unroll
    for (int jk = 0; jk < 4; ++jk) {
      float bhv = bh[lrow * 32 + kt * 2 + (jk >> 1)];
      const float* bwp = bw + lrow * 32 + (jk & 1) * 16 + 8 * hf;
      v4f b0 = *(const v4f*)bwp;
      v4f b1 = *(const v4f*)(bwp + 4);
      v8f bw8 = __builtin_shufflevector(b0, b1, 0,1,2,3,4,5,6,7);
      v8bf mv = *(const v8bf*)(mb + kt * 64 + jk * 16 + 8 * hf);
      v8f mv8 = __builtin_convertvector(mv, v8f);
      S[jk] = S[jk] + bw8 + mv8 + bhv;
    }

    // online softmax: per-lane scalar state, single cross-half shuffle
    v8f t8 = __builtin_elementwise_max(__builtin_elementwise_max(S[0], S[1]),
                                       __builtin_elementwise_max(S[2], S[3]));
    float t = t8[0];
    #pragma unroll
    for (int r = 1; r < 8; ++r) t = fmaxf(t, t8[r]);
    t = fmaxf(t, __shfl_xor(t, 16, 32));
    float nmax = fmaxf(mcur, t);
    float sc = __builtin_amdgcn_exp2f((mcur - nmax) * LOG2E);
    #pragma unroll
    for (int jd = 0; jd < 4; ++jd) O[jd] = O[jd] * sc;
    #pragma unroll
    for (int jk = 0; jk < 4; ++jk)
      #pragma unroll
      for (int r = 0; r < 8; ++r)
        S[jk][r] = __builtin_amdgcn_exp2f((S[jk][r] - nmax) * LOG2E);
    v8f acc8 = (S[0] + S[1]) + (S[2] + S[3]);
    float acc = acc8[0];
    #pragma unroll
    for (int r = 1; r < 8; ++r) acc += acc8[r];
    acc += __shfl_xor(acc, 16, 32);
    scur = scur * sc + acc;
    mcur = nmax;

    // assemble P B-fragments in registers (pack + cross-half exchange)
    v16bf Bp[2];
    #pragma unroll
    for (int s = 0; s < 2; ++s) {
      unsigned int pA[4], pB[4];
      #pragma unroll
      for (int u = 0; u < 4; ++u) {
        pA[u] = pack2(S[2 * s][2 * u],     S[2 * s][2 * u + 1]);
        pB[u] = pack2(S[2 * s + 1][2 * u], S[2 * s + 1][2 * u + 1]);
      }
      v4u lo, hi;
      #pragma unroll
      for (int u = 0; u < 4; ++u) {
        unsigned int snd = hf ? pA[u] : pB[u];
        unsigned int rcv = (unsigned int)__shfl_xor((int)snd, 16, 32);
        lo[u] = hf ? rcv   : pA[u];
        hi[u] = hf ? pB[u] : rcv;
      }
      Bp[s] = frag_from(lo, hi);
    }

    // O^T += V^T * P  (A = V^T rows directly from dim-major LDS)
    #pragma unroll
    for (int jd = 0; jd < 4; ++jd) {
      const bf16* vr = Vc + (jd * 16 + ln) * ST;
      O[jd] = wmma_bf(frag_a(vr, 0, hf), Bp[0], O[jd]);
      O[jd] = wmma_bf(frag_a(vr, 32, hf), Bp[1], O[jd]);
    }

    // drain prefetched tile into the other buffer
    if (more) {
      #pragma unroll
      for (int u = 0; u < 8; ++u) {
        int i = tid + u * 128;
        int row = i >> 4, c4 = i & 15;
        v4f f = kst[u];
        v4bf p;
        p[0] = (bf16)f[0]; p[1] = (bf16)f[1]; p[2] = (bf16)f[2]; p[3] = (bf16)f[3];
        *(v4bf*)(Kn + row * ST + c4 * 4) = p;
      }
      #pragma unroll
      for (int wi = 0; wi < 2; ++wi) {
        int k0 = vk0 + wi * 32;
        #pragma unroll
        for (int d = 0; d < 4; ++d) {
          v4bf pv;
          pv[0] = (bf16)vst[wi][0][d]; pv[1] = (bf16)vst[wi][1][d];
          pv[2] = (bf16)vst[wi][2][d]; pv[3] = (bf16)vst[wi][3][d];
          *(v4bf*)(Vn + (vc4 * 4 + d) * ST + k0) = pv;
        }
      }
    }
    __syncthreads();
  }

  // ---- epilogue: normalize, contiguous b128 stores of this lane's query row ----
  float inv = __builtin_amdgcn_rcpf(scur);
  int h = grow >> 5, w = grow & 31;
  float* orow = outg + (((size_t)batch * 32 + h) * 32 + w) * 512 + head * 64;
  #pragma unroll
  for (int jd = 0; jd < 4; ++jd) {
    int d0 = jd * 16 + 8 * hf;
    v8f ov = O[jd] * inv;
    v4f lo = __builtin_shufflevector(ov, ov, 0, 1, 2, 3);
    v4f hi = __builtin_shufflevector(ov, ov, 4, 5, 6, 7);
    *(v4f*)(orow + d0) = lo;
    *(v4f*)(orow + d0 + 4) = hi;
  }
}

extern "C" void kernel_launch(void* const* d_in, const int* in_sizes, int n_in,
                              void* d_out, int out_size, void* d_ws, size_t ws_size,
                              hipStream_t stream) {
  (void)in_sizes; (void)n_in; (void)out_size; (void)d_ws; (void)ws_size;
  const float* q = (const float*)d_in[0];
  const float* k = (const float*)d_in[1];
  const float* v = (const float*)d_in[2];
  const unsigned char* mask = (const unsigned char*)d_in[3];
  const float* relw = (const float*)d_in[4];
  const float* relh = (const float*)d_in[5];
  float* out = (float*)d_out;

  dim3 grid(8 * 8 * 16);   // batch x heads x 16 query tiles of 64 rows
  dim3 block(128);         // 4 waves, one 16-query WMMA stripe each
  relpos_attn_wmma<<<grid, block, 0, stream>>>(q, k, v, mask, relw, relh, out);
}
